// cross_spatial_qk2v_k_relu_33621003993166
// MI455X (gfx1250) — matrix operand order repaired and checked by reading, not compiled, not run on hardware
//
#include <hip/hip_runtime.h>

typedef __attribute__((ext_vector_type(16))) _Float16 v16h;
typedef __attribute__((ext_vector_type(8)))  _Float16 v8h;
typedef __attribute__((ext_vector_type(4)))  _Float16 v4h;
typedef __attribute__((ext_vector_type(8)))  float    v8f;
typedef __attribute__((ext_vector_type(4)))  float    v4f;

#define BDIM 8
#define SEQ  1024
#define CH   512
#define NH   8
#define HD   64
#define SCALE_F 0.125f

#define LDK 72   // padded LDS row stride (halves) for 64-wide tiles

__device__ __forceinline__ v16h cat8(v8h a, v8h b) {
  return __builtin_shufflevector(a, b, 0,1,2,3,4,5,6,7,8,9,10,11,12,13,14,15);
}

__device__ __forceinline__ v8f wmma32(v16h a, v16h b, v8f c) {
  return __builtin_amdgcn_wmma_f32_16x16x32_f16(false, a, false, b, (short)0, c, false, false);
}

// CDNA5 async copy: 16B global -> LDS per lane, tracked by ASYNCcnt.
// ldsoff = wave-relative LDS byte offset (low 32 bits of generic pointer),
// voff   = per-lane byte offset added to the uniform 64-bit SGPR base.
__device__ __forceinline__ void async_cp16(const void* gbase, unsigned voff, unsigned ldsoff) {
  asm volatile("global_load_async_to_lds_b128 %0, %1, %2"
               :: "v"(ldsoff), "v"(voff), "s"(gbase) : "memory");
}
__device__ __forceinline__ void wait_async_le4() {
  asm volatile("s_wait_asynccnt 0x4" ::: "memory");
}
__device__ __forceinline__ void wait_async_0() {
  asm volatile("s_wait_asynccnt 0x0" ::: "memory");
}
__device__ __forceinline__ unsigned lds_off(const void* p) {
  return (unsigned)(uintptr_t)p;   // generic LDS addr: low 32 bits = LDS offset
}

// ---------------------------------------------------------------------------
// Phase 1: kv = x @ W for both branches. Writes K (f16, [br][b][h][n][d]) and
// V transposed (f16, [br][b][h][d][n]) into workspace.
// Grid: 2 branches x 64 row-blocks(128) x 8 col-blocks(128) = 1024 blocks.
// ---------------------------------------------------------------------------
__global__ __launch_bounds__(256)
void kv_gemm_kernel(const float* __restrict__ x1, const float* __restrict__ x2,
                    const float* __restrict__ W1, const float* __restrict__ W2,
                    _Float16* __restrict__ Kbuf, _Float16* __restrict__ Vt) {
  __shared__ _Float16 Xs[128 * LDK];   // 128 rows x 64 k (padded)
  __shared__ _Float16 Ws[128 * LDK];   // 128 n   x 64 k (W transposed)

  const int bx     = blockIdx.x;
  const int branch = bx >> 9;
  const int rem    = bx & 511;
  const int rb     = rem >> 3;   // 0..63
  const int cb     = rem & 7;    // 0..7
  const float* X = branch ? x2 : x1;
  const float* W = branch ? W2 : W1;

  const int tid = threadIdx.x;
  const int w   = tid >> 5;
  const int l   = tid & 31;
  const int ln  = l & 15;
  const int hh  = l >> 4;

  v8f acc[8] = {};

  for (int kt = 0; kt < CH; kt += 64) {
    // X tile: 128x64 f32 -> f16 (row-major)
    #pragma unroll
    for (int it = 0; it < 8; ++it) {
      int e = it * 256 + tid;            // float4 unit, 2048 total
      int r = e >> 4;
      int c = (e & 15) << 2;
      v4f f = *(const v4f*)(X + (size_t)(rb * 128 + r) * CH + kt + c);
      v4h hq;
      hq[0] = (_Float16)f[0]; hq[1] = (_Float16)f[1];
      hq[2] = (_Float16)f[2]; hq[3] = (_Float16)f[3];
      *(v4h*)(&Xs[r * LDK + c]) = hq;
    }
    // W tile: 64k x 128n f32 -> f16 transposed into Ws[n][k]
    #pragma unroll
    for (int it = 0; it < 8; ++it) {
      int e = it * 256 + tid;
      int k = e >> 5;
      int n = (e & 31) << 2;
      v4f f = *(const v4f*)(W + (size_t)(kt + k) * (2 * CH) + cb * 128 + n);
      #pragma unroll
      for (int j = 0; j < 4; ++j) Ws[(n + j) * LDK + k] = (_Float16)f[j];
    }
    __syncthreads();

    const _Float16* xrow = &Xs[(16 * w + ln) * LDK];
    v16h a0 = cat8(*(const v8h*)(xrow +      16 * hh), *(const v8h*)(xrow + 8 + 16 * hh));
    v16h a1 = cat8(*(const v8h*)(xrow + 32 + 16 * hh), *(const v8h*)(xrow + 40 + 16 * hh));
    #pragma unroll
    for (int nt = 0; nt < 8; ++nt) {
      const _Float16* wrow = &Ws[(16 * nt + ln) * LDK + 16 * hh];
      v16h b0 = cat8(*(const v8h*)(wrow),      *(const v8h*)(wrow + 8));
      v16h b1 = cat8(*(const v8h*)(wrow + 32), *(const v8h*)(wrow + 40));
      acc[nt] = wmma32(a0, b0, acc[nt]);
      acc[nt] = wmma32(a1, b1, acc[nt]);
    }
    __syncthreads();
  }

  // scatter-store to K / V^T (f16)
  const int rowbase = rb * 128 + 16 * w;
  #pragma unroll
  for (int nt = 0; nt < 8; ++nt) {
    int col = cb * 128 + 16 * nt + ln;   // 0..1023 over 2C
    #pragma unroll
    for (int r = 0; r < 8; ++r) {
      int rg = rowbase + r + 8 * hh;     // 0..8191
      int b  = rg >> 10;
      int nn = rg & 1023;
      _Float16 hv = (_Float16)acc[nt][r];
      if (col < CH) {                    // K half (uniform per block)
        int hI = col >> 6, d = col & 63;
        Kbuf[((((size_t)branch * BDIM + b) * NH + hI) * SEQ + nn) * HD + d] = hv;
      } else {                           // V half -> transposed
        int c2 = col - CH;
        int hI = c2 >> 6, d = c2 & 63;
        Vt[((((size_t)branch * BDIM + b) * NH + hI) * HD + d) * SEQ + nn] = hv;
      }
    }
  }
}

// ---------------------------------------------------------------------------
// Phase 2: fused relu-attention, double-buffered async LDS staging.
// out = relu(Q Kx^T * s) @ V accumulated over 64-key chunks.
// Grid: 2 branches x (B*H=64) x 8 q-blocks = 1024 blocks.
// branch 0: Q=x1, K from kv2, V from kv1.  branch 1: Q=x2, K from kv1, V=kv2.
// ---------------------------------------------------------------------------
#define NCHUNK 16

__global__ __launch_bounds__(256)
void attn_kernel(const float* __restrict__ x1, const float* __restrict__ x2,
                 const _Float16* __restrict__ Kbuf, const _Float16* __restrict__ Vt,
                 float* __restrict__ out) {
  __shared__ _Float16 Ks[2][64 * LDK];     // 64 keys x 64 d   (double buffered)
  __shared__ _Float16 Vs[2][64 * LDK];     // 64 d    x 64 keys (V^T, dbl buf)
  __shared__ _Float16 Sm[8 * 16 * LDK];    // per-wave 16 x 64 score strip

  const int bx     = blockIdx.x;
  const int branch = bx >> 9;
  const int rem    = bx & 511;
  const int bh     = rem >> 3;
  const int qb     = rem & 7;
  const int b      = bh >> 3;
  const int head   = bh & 7;

  const float* X = branch ? x2 : x1;
  const _Float16* Kp = Kbuf + (((size_t)(1 - branch) * BDIM + b) * NH + head) * (size_t)(SEQ * HD);
  const _Float16* Vp = Vt   + (((size_t)branch       * BDIM + b) * NH + head) * (size_t)(HD * SEQ);
  float* outp = out + (size_t)branch * BDIM * SEQ * CH;

  const int tid = threadIdx.x;
  const int w   = tid >> 5;
  const int l   = tid & 31;
  const int ln  = l & 15;
  const int hh  = l >> 4;

  // per-thread staging coordinates (2 x 16B units each for K and V)
  const int u0 = tid, u1 = tid + 256;
  const int kr0 = u0 >> 3, kc0 = (u0 & 7) << 3;
  const int kr1 = u1 >> 3, kc1 = (u1 & 7) << 3;
  const unsigned kvoff0 = (unsigned)u0 * 16u;            // bytes into 64x64 chunk
  const unsigned kvoff1 = (unsigned)u1 * 16u;
  const unsigned vvoff0 = (unsigned)((kr0 * SEQ + kc0) * 2);  // V^T: row stride SEQ
  const unsigned vvoff1 = (unsigned)((kr1 * SEQ + kc1) * 2);

  // Q A-fragments (16 rows x 64 d), f32 global -> f16, WMMA A layout
  const int qrow = qb * 128 + 16 * w + ln;
  const float* qp = X + ((size_t)b * SEQ + qrow) * CH + head * HD;
  v16h qA[2];
  #pragma unroll
  for (int kk = 0; kk < 2; ++kk) {
    v4f f0 = *(const v4f*)(qp + kk * 32 + 8 * hh);
    v4f f1 = *(const v4f*)(qp + kk * 32 + 8 * hh + 4);
    v4f f2 = *(const v4f*)(qp + kk * 32 + 16 + 8 * hh);
    v4f f3 = *(const v4f*)(qp + kk * 32 + 16 + 8 * hh + 4);
    v8h g0, g1;
    #pragma unroll
    for (int j = 0; j < 4; ++j) {
      g0[j] = (_Float16)f0[j]; g0[4 + j] = (_Float16)f1[j];
      g1[j] = (_Float16)f2[j]; g1[4 + j] = (_Float16)f3[j];
    }
    qA[kk] = cat8(g0, g1);
  }

  v8f acc[4] = {};
  _Float16* Sw = &Sm[w * 16 * LDK];

  // prologue: async-stage chunk 0 into buffer 0
  {
    const void* kb = (const void*)Kp;            // m0 = 0
    const void* vb = (const void*)Vp;
    async_cp16(kb, kvoff0, lds_off(&Ks[0][kr0 * LDK + kc0]));
    async_cp16(kb, kvoff1, lds_off(&Ks[0][kr1 * LDK + kc1]));
    async_cp16(vb, vvoff0, lds_off(&Vs[0][kr0 * LDK + kc0]));
    async_cp16(vb, vvoff1, lds_off(&Vs[0][kr1 * LDK + kc1]));
  }

  for (int mc = 0; mc < NCHUNK; ++mc) {
    const int cur = mc & 1;
    // prefetch next chunk into the other buffer, then wait for current chunk
    if (mc + 1 < NCHUNK) {
      const int nxt = (mc + 1) & 1;
      const int m1 = (mc + 1) * 64;
      const void* kb = (const void*)(Kp + (size_t)m1 * HD);
      const void* vb = (const void*)(Vp + m1);
      async_cp16(kb, kvoff0, lds_off(&Ks[nxt][kr0 * LDK + kc0]));
      async_cp16(kb, kvoff1, lds_off(&Ks[nxt][kr1 * LDK + kc1]));
      async_cp16(vb, vvoff0, lds_off(&Vs[nxt][kr0 * LDK + kc0]));
      async_cp16(vb, vvoff1, lds_off(&Vs[nxt][kr1 * LDK + kc1]));
      wait_async_le4();                  // in-order: current chunk's 4 are done
    } else {
      wait_async_0();
    }
    __syncthreads();

    // S = Q K^T, relu(scale*S), park as f16 in per-wave LDS strip
    #pragma unroll
    for (int nt = 0; nt < 4; ++nt) {
      const _Float16* krow = &Ks[cur][(16 * nt + ln) * LDK + 8 * hh];
      v16h b0 = cat8(*(const v8h*)(krow),      *(const v8h*)(krow + 16));
      v16h b1 = cat8(*(const v8h*)(krow + 32), *(const v8h*)(krow + 48));
      v8f s = {};
      s = wmma32(qA[0], b0, s);
      s = wmma32(qA[1], b1, s);
      #pragma unroll
      for (int r = 0; r < 8; ++r) {
        float vsc = s[r] * SCALE_F;
        vsc = vsc > 0.0f ? vsc : 0.0f;
        Sw[(r + 8 * hh) * LDK + 16 * nt + ln] = (_Float16)vsc;
      }
    }
    // out += S @ V  (wave-private Sw: no cross-wave dependency)
    #pragma unroll
    for (int kc = 0; kc < 2; ++kc) {
      const _Float16* srow = &Sw[ln * LDK + kc * 32 + 16 * hh];
      v16h a2 = cat8(*(const v8h*)(srow), *(const v8h*)(srow + 8));
      #pragma unroll
      for (int dt = 0; dt < 4; ++dt) {
        const _Float16* vrow = &Vs[cur][(16 * dt + ln) * LDK + kc * 32 + 16 * hh];
        v16h b2 = cat8(*(const v8h*)(vrow), *(const v8h*)(vrow + 8));
        acc[dt] = wmma32(a2, b2, acc[dt]);
      }
    }
    __syncthreads();
  }

  // store f32 output (b, n, h*64+d)
  #pragma unroll
  for (int dt = 0; dt < 4; ++dt) {
    #pragma unroll
    for (int r = 0; r < 8; ++r) {
      int grow = qb * 128 + 16 * w + r + 8 * hh;
      outp[((size_t)b * SEQ + grow) * CH + head * HD + 16 * dt + ln] = acc[dt][r];
    }
  }
}

extern "C" void kernel_launch(void* const* d_in, const int* in_sizes, int n_in,
                              void* d_out, int out_size, void* d_ws, size_t ws_size,
                              hipStream_t stream) {
  const float* x1 = (const float*)d_in[0];
  const float* x2 = (const float*)d_in[1];
  const float* W1 = (const float*)d_in[2];
  const float* W2 = (const float*)d_in[3];

  _Float16* Kbuf = (_Float16*)d_ws;
  _Float16* Vt   = Kbuf + (size_t)2 * BDIM * NH * SEQ * HD;   // +8.4M halves

  kv_gemm_kernel<<<1024, 256, 0, stream>>>(x1, x2, W1, W2, Kbuf, Vt);
  attn_kernel  <<<1024, 256, 0, stream>>>(x1, x2, Kbuf, Vt, (float*)d_out);
}
